// Network_20151986553470
// MI455X (gfx1250) — matrix operand order repaired and checked by reading, not compiled, not run on hardware
//
#include <hip/hip_runtime.h>
#include <hip/hip_bf16.h>

typedef _Float16 v16h __attribute__((ext_vector_type(16)));
typedef _Float16 v8h  __attribute__((ext_vector_type(8)));
typedef float    v8f  __attribute__((ext_vector_type(8)));
typedef unsigned int u32x4 __attribute__((ext_vector_type(4)));
typedef int          i32x4 __attribute__((ext_vector_type(4)));
typedef int          i32x8 __attribute__((ext_vector_type(8)));

#define WMMA_F16(A,B,C) __builtin_amdgcn_wmma_f32_16x16x32_f16(false,(A),false,(B),(short)0,(C),false,false)

#define KN_E 4096
#define KN_R 16

#if defined(__has_builtin)
#if __has_builtin(__builtin_amdgcn_tensor_load_to_lds)
#define KN_USE_TDM 1
#endif
#endif

// ---------------------------------------------------------------------------
// Pass 1: expert id + positional encodings (f16, padded) + per-expert counts
// ---------------------------------------------------------------------------
__global__ void kilonerf_prep(const float* __restrict__ pts,
                              const float* __restrict__ dirs,
                              int N, int samples,
                              int* __restrict__ eIds, int* __restrict__ counts,
                              _Float16* __restrict__ xpB, _Float16* __restrict__ xdB)
{
    int i = blockIdx.x * blockDim.x + threadIdx.x;
    if (i >= N) return;

    float px = pts[3*i+0], py = pts[3*i+1], pz = pts[3*i+2];
    const float scale = (float)KN_R / 3.0f;           // R / (AABB_MAX-AABB_MIN)
    float fx = fminf(fmaxf((px + 1.5f) * scale, 0.f), (float)(KN_R-1));
    float fy = fminf(fmaxf((py + 1.5f) * scale, 0.f), (float)(KN_R-1));
    float fz = fminf(fmaxf((pz + 1.5f) * scale, 0.f), (float)(KN_R-1));
    int ix = (int)fx, iy = (int)fy, iz = (int)fz;
    int e = ix * (KN_R*KN_R) + iy * KN_R + iz;
    eIds[i] = e;
    atomicAdd(&counts[e], 1);

    _Float16* xp = xpB + (size_t)i * 64;
    xp[0] = (_Float16)px; xp[1] = (_Float16)py; xp[2] = (_Float16)pz;
    int o = 3; float f = 1.0f;
    #pragma unroll
    for (int l = 0; l < 10; ++l) {
        xp[o+0] = (_Float16)sinf(f*px); xp[o+1] = (_Float16)sinf(f*py); xp[o+2] = (_Float16)sinf(f*pz);
        xp[o+3] = (_Float16)cosf(f*px); xp[o+4] = (_Float16)cosf(f*py); xp[o+5] = (_Float16)cosf(f*pz);
        o += 6; f *= 2.0f;
    }
    xp[63] = (_Float16)0.f;

    int ray = i / samples;
    float dx = dirs[3*ray+0], dy = dirs[3*ray+1], dz = dirs[3*ray+2];
    _Float16* xd = xdB + (size_t)i * 32;
    xd[0] = (_Float16)dx; xd[1] = (_Float16)dy; xd[2] = (_Float16)dz;
    o = 3; f = 1.0f;
    #pragma unroll
    for (int l = 0; l < 4; ++l) {
        xd[o+0] = (_Float16)sinf(f*dx); xd[o+1] = (_Float16)sinf(f*dy); xd[o+2] = (_Float16)sinf(f*dz);
        xd[o+3] = (_Float16)cosf(f*dx); xd[o+4] = (_Float16)cosf(f*dy); xd[o+5] = (_Float16)cosf(f*dz);
        o += 6; f *= 2.0f;
    }
    #pragma unroll
    for (int j = 27; j < 32; ++j) xd[j] = (_Float16)0.f;
}

// ---------------------------------------------------------------------------
// Pass 2: exclusive scan of 4096 counts (single block, 256 threads x 16)
// ---------------------------------------------------------------------------
__global__ void kilonerf_scan(const int* __restrict__ counts,
                              int* __restrict__ offs, int* __restrict__ cursor)
{
    __shared__ int sm[256];
    int t = threadIdx.x;
    int loc[16]; int s = 0;
    #pragma unroll
    for (int j = 0; j < 16; ++j) { int v = counts[t*16 + j]; loc[j] = s; s += v; }
    sm[t] = s;
    __syncthreads();
    int mySum = s;
    for (int off = 1; off < 256; off <<= 1) {
        int v = (t >= off) ? sm[t - off] : 0;
        __syncthreads();
        sm[t] += v;
        __syncthreads();
    }
    int prefix = sm[t] - mySum;
    #pragma unroll
    for (int j = 0; j < 16; ++j) {
        int o = prefix + loc[j];
        offs[t*16 + j] = o;
        cursor[t*16 + j] = o;
    }
}

// ---------------------------------------------------------------------------
// Pass 3: scatter point ids into per-expert contiguous lists
// ---------------------------------------------------------------------------
__global__ void kilonerf_scatter(const int* __restrict__ eIds,
                                 int* __restrict__ cursor, int* __restrict__ list, int N)
{
    int i = blockIdx.x * blockDim.x + threadIdx.x;
    if (i >= N) return;
    int e = eIds[i];
    int pos = atomicAdd(&cursor[e], 1);
    list[pos] = i;
}

// ---------------------------------------------------------------------------
// TDM helper: 1D contiguous f32 block, global -> LDS (D# per ISA 8.3/8.4).
// ---------------------------------------------------------------------------
#ifdef KN_USE_TDM
static __device__ __forceinline__ void tdm_load_1d(unsigned int lds_off,
                                                   const void* gptr,
                                                   unsigned int nElems)
{
    unsigned long long ga = (unsigned long long)(uintptr_t)gptr;
    u32x4 g0;
    g0[0] = 1u;                                       // count=1, user descriptor
    g0[1] = lds_off;                                  // lds_addr (bytes)
    g0[2] = (unsigned int)ga;                         // global_addr[31:0]
    g0[3] = (unsigned int)((ga >> 32) & 0x1FFFFFFu)   // global_addr[56:32]
          | (2u << 30);                               // type=2 ("image")
    i32x8 g1;
    g1[0] = (int)(2u << 16);                          // wg_mask=0, data_size=4B
    g1[1] = (int)((nElems & 0xFFFFu) << 16);          // tensor_dim0 lo16
    g1[2] = (int)(((nElems >> 16) & 0xFFFFu)          // tensor_dim0 hi16
          | (1u << 16));                              // tensor_dim1 = 1
    g1[3] = (int)((nElems & 0xFFFFu) << 16);          // tile_dim0 = nElems
    g1[4] = 0;                                        // tile_dim1/2 unused
    g1[5] = (int)nElems;                              // tensor_dim0_stride lo
    g1[6] = 0;                                        // stride hi / dim1_stride lo
    g1[7] = 0;                                        // dim1_stride hi
    i32x4 z4 = {0, 0, 0, 0};                          // groups 2/3: dims unused
#if defined(__clang_major__) && (__clang_major__ >= 23)
    i32x8 z8 = {0, 0, 0, 0, 0, 0, 0, 0};
    __builtin_amdgcn_tensor_load_to_lds(g0, g1, z4, z4, z8, 0);
#else
    __builtin_amdgcn_tensor_load_to_lds(g0, g1, z4, z4, 0);
#endif
}
#endif

// ---------------------------------------------------------------------------
// Pass 4: per-expert WMMA MLP. 1 workgroup (4 waves) per expert.
// Raw f32 weights DMA'd to LDS via TDM, converted once to f16 fragments:
//   frag f occupies wlds[f*512 .. f*512+511]; lane reads 16 contiguous halves.
// B layout (32x16 f16): lane L -> col N=L&15; g=L>>4: halves 0-7 = K g*8.., halves 8-15 = K 16+g*8...
// A layout (16x32 f16): lane L -> row M=L&15; g=L>>4: halves 0-7 = K g*8..,
//   halves 8-15 = K 16+g*8...
// ---------------------------------------------------------------------------
static __device__ __forceinline__ v16h load_a_row(const _Float16* row, int o)
{
    v8h lo = *(const v8h*)(row + o);
    v8h hi = *(const v8h*)(row + o + 16);
    v16h a;
    #pragma unroll
    for (int i = 0; i < 8; ++i) { a[i] = lo[i]; a[i+8] = hi[i]; }
    return a;
}

// raw staging offsets (floats)
#define O_W1   0
#define O_W2   2016
#define O_WF   3040
#define O_WS   4064
#define O_WV   4096
#define O_WRGB 5984
#define O_RAW_TOTAL 6080

__global__ void __launch_bounds__(128) kilonerf_mlp(
    const float* __restrict__ W1,  const float* __restrict__ b1,
    const float* __restrict__ W2,  const float* __restrict__ b2,
    const float* __restrict__ Wf,  const float* __restrict__ bf,
    const float* __restrict__ Wsig,const float* __restrict__ bsig,
    const float* __restrict__ Wv,  const float* __restrict__ bv,
    const float* __restrict__ Wrgb,const float* __restrict__ brgb,
    const int* __restrict__ list, const int* __restrict__ counts, const int* __restrict__ offs,
    const _Float16* __restrict__ xpB, const _Float16* __restrict__ xdB,
    float* __restrict__ rgbOut, float* __restrict__ sigOut)
{
    int e = blockIdx.x;
    int nPts = counts[e];
    if (nPts == 0) return;
    int base = offs[e];

    // Frag map: 0-3 W1(kt=f>>1,nt=f&1) | 4-5 W2 | 6-7 Wf | 8 Wsig | 9-12 Wv | 13 Wrgb
    __shared__ __align__(32) _Float16 wlds[14 * 512];
    __shared__ float blds[132];
    __shared__ __align__(16) _Float16 hbuf[4][16 * 64];
#ifdef KN_USE_TDM
    __shared__ __align__(16) float rawS[O_RAW_TOTAL];
#endif

    int tid = threadIdx.x;
    int lane = tid & 31;
    int wave = tid >> 5;

#ifdef KN_USE_TDM
    // Stage this expert's raw f32 weights into LDS with the Tensor Data Mover.
    if (wave == 0) {
        unsigned int ldsBase = (unsigned int)(uintptr_t)(void*)rawS;
        tdm_load_1d(ldsBase + O_W1  *4, W1   + (size_t)e*63*32, 2016);
        tdm_load_1d(ldsBase + O_W2  *4, W2   + (size_t)e*32*32, 1024);
        tdm_load_1d(ldsBase + O_WF  *4, Wf   + (size_t)e*32*32, 1024);
        tdm_load_1d(ldsBase + O_WS  *4, Wsig + (size_t)e*32,      32);
        tdm_load_1d(ldsBase + O_WV  *4, Wv   + (size_t)e*59*32, 1888);
        tdm_load_1d(ldsBase + O_WRGB*4, Wrgb + (size_t)e*32*3,    96);
        __builtin_amdgcn_s_wait_tensorcnt(0);
    }
#endif
    if (tid < 132) {
        float v;
        if      (tid < 32)  v = b1[e*32 + tid];
        else if (tid < 64)  v = b2[e*32 + (tid-32)];
        else if (tid < 96)  v = bf[e*32 + (tid-64)];
        else if (tid == 96) v = bsig[e];
        else if (tid < 129) v = bv[e*32 + (tid-97)];
        else                v = brgb[e*3 + (tid-129)];
        blds[tid] = v;
    }
    __syncthreads();

    // Convert raw weights into f16 B-fragment layout in LDS.
    for (int idx = tid; idx < 14 * 512; idx += 128) {
        int frag = idx >> 9;
        int rem  = idx & 511;
        int ln   = rem >> 4;
        int h    = rem & 15;
        int cc = ln & 15, gg = ln >> 4;
        int kt, nt;
        if      (frag < 4)  { kt = frag >> 1; nt = frag & 1; }
        else if (frag < 6)  { kt = 0; nt = frag - 4; }
        else if (frag < 8)  { kt = 0; nt = frag - 6; }
        else if (frag == 8) { kt = 0; nt = 0; }
        else if (frag < 13) { int q = frag - 9; kt = q >> 1; nt = q & 1; }
        else                { kt = 0; nt = 0; }
        int k = kt*32 + ((h < 8) ? (gg*8 + h) : (16 + gg*8 + (h - 8)));
        int n = nt*16 + cc;
        float v;
#ifdef KN_USE_TDM
        if      (frag < 4)  v = (k < 63) ? rawS[O_W1 + k*32 + n] : 0.f;
        else if (frag < 6)  v = rawS[O_W2 + k*32 + n];
        else if (frag < 8)  v = rawS[O_WF + k*32 + n];
        else if (frag == 8) v = (n == 0) ? rawS[O_WS + k] : 0.f;
        else if (frag < 13) v = (k < 59) ? rawS[O_WV + k*32 + n] : 0.f;
        else                v = (n < 3) ? rawS[O_WRGB + k*3 + n] : 0.f;
#else
        if      (frag < 4)  v = (k < 63) ? W1[((size_t)e*63 + k)*32 + n] : 0.f;
        else if (frag < 6)  v = W2[((size_t)e*32 + k)*32 + n];
        else if (frag < 8)  v = Wf[((size_t)e*32 + k)*32 + n];
        else if (frag == 8) v = (n == 0) ? Wsig[(size_t)e*32 + k] : 0.f;
        else if (frag < 13) v = (k < 59) ? Wv[((size_t)e*59 + k)*32 + n] : 0.f;
        else                v = (n < 3) ? Wrgb[((size_t)e*32 + k)*3 + n] : 0.f;
#endif
        wlds[idx] = (_Float16)v;
    }
    __syncthreads();

    int c = lane & 15, g = lane >> 4;
    _Float16* hb = hbuf[wave];
    const _Float16* wl = wlds;
    int numTiles = (nPts + 15) >> 4;

    #define BFRAG(f) (*(const v16h*)(wl + (f)*512 + lane*16))

    for (int t = wave; t < numTiles; t += 4) {
        int rowBase = base + t*16;
        int cnt = nPts - t*16; if (cnt > 16) cnt = 16;
        int mA = c;
        bool validA = mA < cnt;
        int ptA = list[rowBase + (validA ? mA : 0)];

        const _Float16* xpRow = xpB + (size_t)ptA * 64;
        v16h a0 = load_a_row(xpRow, g*8);          // K 0..31
        v16h a1 = load_a_row(xpRow, 32 + g*8);     // K 32..63
        if (!validA) { v16h z = {}; a0 = z; a1 = z; }

        // ---- layer 1: 63 -> 32, relu ----
        v8f h0 = {}; v8f h1 = {};
        h0 = WMMA_F16(a0, BFRAG(0), h0);
        h0 = WMMA_F16(a1, BFRAG(2), h0);
        h1 = WMMA_F16(a0, BFRAG(1), h1);
        h1 = WMMA_F16(a1, BFRAG(3), h1);
        float bb0 = blds[c], bb1 = blds[16 + c];
        #pragma unroll
        for (int r = 0; r < 8; ++r) {
            int m = r + 8*g;
            hb[m*64 + c]      = (_Float16)fmaxf(h0[r] + bb0, 0.f);
            hb[m*64 + 16 + c] = (_Float16)fmaxf(h1[r] + bb1, 0.f);
        }
        asm volatile("s_wait_dscnt 0x0" ::: "memory");
        v16h ah = load_a_row(hb + mA*64, g*8);

        // ---- layer 2: 32 -> 32, relu ----
        v8f h20 = {}; v8f h21 = {};
        h20 = WMMA_F16(ah, BFRAG(4), h20);
        h21 = WMMA_F16(ah, BFRAG(5), h21);
        bb0 = blds[32 + c]; bb1 = blds[48 + c];
        #pragma unroll
        for (int r = 0; r < 8; ++r) {
            int m = r + 8*g;
            hb[m*64 + c]      = (_Float16)fmaxf(h20[r] + bb0, 0.f);
            hb[m*64 + 16 + c] = (_Float16)fmaxf(h21[r] + bb1, 0.f);
        }
        asm volatile("s_wait_dscnt 0x0" ::: "memory");
        v16h ah2 = load_a_row(hb + mA*64, g*8);

        // ---- sigma: 32 -> 1 (col 0 of frag 8) ----
        v8f sAcc = {};
        sAcc = WMMA_F16(ah2, BFRAG(8), sAcc);

        // ---- feat: 32 -> 32, no relu ----
        v8f f0 = {}; v8f f1 = {};
        f0 = WMMA_F16(ah2, BFRAG(6), f0);
        f1 = WMMA_F16(ah2, BFRAG(7), f1);
        bb0 = blds[64 + c]; bb1 = blds[80 + c];
        #pragma unroll
        for (int r = 0; r < 8; ++r) {
            int m = r + 8*g;
            hb[m*64 + c]      = (_Float16)(f0[r] + bb0);
            hb[m*64 + 16 + c] = (_Float16)(f1[r] + bb1);
        }
        asm volatile("s_wait_dscnt 0x0" ::: "memory");
        v16h af = load_a_row(hb + mA*64, g*8);           // hv K 0..31 (feat)
        v16h ax = load_a_row(xdB + (size_t)ptA*32, g*8); // hv K 32..63 (xd)
        if (!validA) { v16h z = {}; ax = z; }

        // ---- view layer: 59 -> 32, relu ----
        v8f v0 = {}; v8f v1 = {};
        v0 = WMMA_F16(af, BFRAG(9),  v0);
        v0 = WMMA_F16(ax, BFRAG(11), v0);
        v1 = WMMA_F16(af, BFRAG(10), v1);
        v1 = WMMA_F16(ax, BFRAG(12), v1);
        bb0 = blds[97 + c]; bb1 = blds[97 + 16 + c];
        #pragma unroll
        for (int r = 0; r < 8; ++r) {
            int m = r + 8*g;
            hb[m*64 + c]      = (_Float16)fmaxf(v0[r] + bb0, 0.f);
            hb[m*64 + 16 + c] = (_Float16)fmaxf(v1[r] + bb1, 0.f);
        }
        asm volatile("s_wait_dscnt 0x0" ::: "memory");
        v16h ahv = load_a_row(hb + mA*64, g*8);

        // ---- rgb: 32 -> 3 (cols 0..2 of frag 13) ----
        v8f rAcc = {};
        rAcc = WMMA_F16(ahv, BFRAG(13), rAcc);

        float bsigV = blds[96];
        #pragma unroll
        for (int r = 0; r < 8; ++r) {
            int m = r + 8*g;
            if (m < cnt) {
                int pt = list[rowBase + m];
                if (c < 3)  rgbOut[(size_t)pt*3 + c] = rAcc[r] + blds[129 + c];
                if (c == 0) sigOut[pt] = sAcc[r] + bsigV;
            }
        }
    }
    #undef BFRAG
}

// ---------------------------------------------------------------------------
extern "C" void kernel_launch(void* const* d_in, const int* in_sizes, int n_in,
                              void* d_out, int out_size, void* d_ws, size_t ws_size,
                              hipStream_t stream)
{
    const float* pts  = (const float*)d_in[0];
    const float* dirs = (const float*)d_in[1];
    const float* W1   = (const float*)d_in[2];
    const float* b1   = (const float*)d_in[3];
    const float* W2   = (const float*)d_in[4];
    const float* b2   = (const float*)d_in[5];
    const float* Wf   = (const float*)d_in[6];
    const float* bf   = (const float*)d_in[7];
    const float* Wsig = (const float*)d_in[8];
    const float* bsig = (const float*)d_in[9];
    const float* Wv   = (const float*)d_in[10];
    const float* bv   = (const float*)d_in[11];
    const float* Wrgb = (const float*)d_in[12];
    const float* brgb = (const float*)d_in[13];

    int N       = in_sizes[0] / 3;   // 65536 points
    int rays    = in_sizes[1] / 3;   // 1024
    int samples = N / rays;          // 64

    // workspace carve (ints first, then 16B-aligned f16 encodings)
    int* wsI     = (int*)d_ws;
    int* counts  = wsI;                         // E
    int* offs    = wsI + KN_E;                  // E
    int* cursor  = wsI + 2*KN_E;                // E
    int* eIds    = wsI + 3*KN_E;                // N
    int* list    = wsI + 3*KN_E + N;            // N
    size_t halfOff = ((size_t)(3*KN_E + 2*N) * 4 + 15) & ~(size_t)15;
    _Float16* xpB = (_Float16*)((char*)d_ws + halfOff);   // N*64 halves
    _Float16* xdB = xpB + (size_t)N * 64;                 // N*32 halves

    float* rgbOut = (float*)d_out;
    float* sigOut = (float*)d_out + (size_t)N * 3;

    hipMemsetAsync(counts, 0, KN_E * sizeof(int), stream);

    int blocks = (N + 255) / 256;
    kilonerf_prep<<<blocks, 256, 0, stream>>>(pts, dirs, N, samples, eIds, counts, xpB, xdB);
    kilonerf_scan<<<1, 256, 0, stream>>>(counts, offs, cursor);
    kilonerf_scatter<<<blocks, 256, 0, stream>>>(eIds, cursor, list, N);
    kilonerf_mlp<<<KN_E, 128, 0, stream>>>(W1, b1, W2, b2, Wf, bf, Wsig, bsig,
                                           Wv, bv, Wrgb, brgb,
                                           list, counts, offs, xpB, xdB,
                                           rgbOut, sigOut);
}
